// def_att_3_13864154431695
// MI455X (gfx1250) — matrix operand order repaired and checked by reading, not compiled, not run on hardware
//
#include <hip/hip_runtime.h>
#include <hip/hip_bf16.h>

// CDNA5 / gfx1250 deformable attention forward.
// bf16 WMMA (f32 accumulate) for the three GEMM stages; f32 VALU for
// LN / offsets / sampling / softmax. wave32 everywhere.
// LDS: A row-major [M][K], B transposed [N][K]; both padded (K+8 halves)
// so every fragment load is two 16B-aligned ds_load_b128, bank-conflict-free.

typedef __attribute__((ext_vector_type(16))) __bf16 v16bf;
typedef __attribute__((ext_vector_type(8)))  __bf16 v8bf;
typedef __attribute__((ext_vector_type(4)))  __bf16 v4bf;
typedef __attribute__((ext_vector_type(8)))  float  v8f;

#define DIM_ 256
#define DPG_ 32
#define LD32 40   // padded pitch for K=32 matrices (halves)
#define LD64 72   // padded pitch for K=64 matrices (halves)

// ---- WMMA fragment helpers (layouts per CDNA5 ISA) -------------
// A (16-bit, 16x32): lane<16 holds M=lane, K = k0+{0..7, 16..23};
//                    lane>=16 holds M=lane-16, K = k0+{8..15, 24..31}.
static __device__ inline v16bf ldsA(const __bf16* p, int m0, int k0, int ld) {
  int lane = threadIdx.x & 31;
  const __bf16* q0 = p + (m0 + (lane & 15)) * ld + k0 + ((lane >> 4) << 3);
  v16bf a;
#pragma unroll
  for (int i = 0; i < 8; ++i) { a[i] = q0[i]; a[i + 8] = q0[i + 16]; }
  return a;
}

// B (16-bit, 32x16): lane holds N = n0+(lane&15), K = k0 + (lane>=16?8:0) + {0..7, 16..23}.
// B is stored TRANSPOSED in LDS as [N][K] -> two runs of 8 contiguous halves per lane.
static __device__ inline v16bf ldsBt(const __bf16* p, int k0, int n0, int ld) {
  int lane = threadIdx.x & 31;
  const __bf16* q0 = p + (n0 + (lane & 15)) * ld + k0 + ((lane >> 4) << 3);
  v16bf b;
#pragma unroll
  for (int i = 0; i < 8; ++i) { b[i] = q0[i]; b[i + 8] = q0[i + 16]; }
  return b;
}

static __device__ inline v8f wmma_bf16(v16bf a, v16bf b, v8f c) {
#if defined(__HIP_DEVICE_COMPILE__)
  return __builtin_amdgcn_wmma_f32_16x16x32_bf16(false, a, false, b,
                                                 (short)0, c, false, false);
#else
  return c;
#endif
}

// ---- K1: LayerNorm over channels + patchify -------------------------------
__global__ __launch_bounds__(256) void k_ln(const float* __restrict__ x,
                                            const float* __restrict__ g,
                                            const float* __restrict__ bta,
                                            float* __restrict__ xp) {
  int pos = blockIdx.x * 256 + threadIdx.x;   // 65536 = B*H*W
  int bb = pos >> 12;
  int hw = pos & 4095;
  int h = hw >> 6, w = hw & 63;
  const float* px = x + (size_t)bb * DIM_ * 4096 + hw;
  float s = 0.f, s2 = 0.f;
  for (int c = 0; c < DIM_; ++c) { float v = px[c * 4096]; s += v; s2 += v * v; }
  float mu = s * (1.f / DIM_);
  float var = s2 * (1.f / DIM_) - mu * mu;
  float rs = rsqrtf(var + 1e-6f);
  int i = h >> 3, p = h & 7, j = w >> 3, qq = w & 7;
  int n = (i * 8 + j) * 16 + bb;
  float* po = xp + (size_t)n * DIM_ * 64 + p * 8 + qq;
  for (int c = 0; c < DIM_; ++c) {
    float v = px[c * 4096];
    po[c * 64] = (v - mu) * rs * g[c] + bta[c];
  }
}

// ---- K0: f32 -> bf16 conversion -------------------------------------------
__global__ __launch_bounds__(256) void k_cvt(const float* __restrict__ src,
                                             __bf16* __restrict__ dst, int ne) {
  int i = blockIdx.x * 256 + threadIdx.x;
  if (i < ne) dst[i] = (__bf16)src[i];
}

// ---- K2: grouped 1x1 to_q via WMMA (64x32 @ 32x64 per (n,g)) --------------
__global__ __launch_bounds__(128) void k_toq(const float* __restrict__ xp,
                                             const float* __restrict__ wq,
                                             float* __restrict__ q) {
  __shared__ __bf16 Ab[64 * LD32];   // wq[g]  [M=o][K=c]
  __shared__ __bf16 Bt[64 * LD32];   // xp^T   [N=s][K=c]
  int ng = blockIdx.x;
  int n = ng >> 3, g = ng & 7;
  int tid = threadIdx.x;
  const float* wg = wq + (size_t)g * 64 * 32;
  for (int i4 = tid * 4; i4 < 2048; i4 += 512) {
    float4 v = *(const float4*)(wg + i4);
    int m = i4 >> 5, c = i4 & 31;
    v4bf o = {(__bf16)v.x, (__bf16)v.y, (__bf16)v.z, (__bf16)v.w};
    *(v4bf*)(Ab + m * LD32 + c) = o;
  }
  const float* xg = xp + ((size_t)n * DIM_ + g * DPG_) * 64;
  for (int i4 = tid * 4; i4 < 2048; i4 += 512) {
    float4 v = *(const float4*)(xg + i4);
    int c = i4 >> 6, s = i4 & 63;
    Bt[(s + 0) * LD32 + c] = (__bf16)v.x;
    Bt[(s + 1) * LD32 + c] = (__bf16)v.y;
    Bt[(s + 2) * LD32 + c] = (__bf16)v.z;
    Bt[(s + 3) * LD32 + c] = (__bf16)v.w;
  }
  __syncthreads();
  int wv_ = tid >> 5, lane = tid & 31;
  int mt = wv_;                                   // 4 waves -> 4 M-tiles
  v16bf a = ldsA(Ab, mt * 16, 0, LD32);
  float* qo = q + (size_t)ng * 64 * 64;
#pragma unroll
  for (int nt = 0; nt < 4; ++nt) {
    v16bf b = ldsBt(Bt, 0, nt * 16, LD32);
    v8f acc; for (int e = 0; e < 8; ++e) acc[e] = 0.f;
    acc = wmma_bf16(a, b, acc);                   // K=32 in one op
    int nn = nt * 16 + (lane & 15);
#pragma unroll
    for (int r = 0; r < 8; ++r) {
      int m = mt * 16 + r + ((lane >> 4) << 3);
      qo[m * 64 + nn] = acc[r];
    }
  }
}

// ---- K3: offset net: dw-conv6 s4 p1 -> GELU -> 1x1 -> tanh*4 -> grid ------
__global__ __launch_bounds__(64) void k_off(const float* __restrict__ q,
                                            const float* __restrict__ w1,
                                            const float* __restrict__ b1,
                                            const float* __restrict__ w2,
                                            float* __restrict__ vgrid) {
  __shared__ float img[64 * 64];
  __shared__ float hb[64 * 4];
  int ng = blockIdx.x;
  int tid = threadIdx.x;
  const float* qs = q + (size_t)ng * 4096;
  for (int idx = tid; idx < 4096; idx += 64) img[idx] = qs[idx];
  __syncthreads();
  const float* row = img + tid * 64;
  const float* wc = w1 + tid * 36;
  float bias = b1[tid];
  for (int oy = 0; oy < 2; ++oy)
    for (int ox = 0; ox < 2; ++ox) {
      float acc = 0.f;
      for (int ky = 0; ky < 6; ++ky) {
        int iy = oy * 4 - 1 + ky;
        if ((unsigned)iy >= 8u) continue;
        for (int kx = 0; kx < 6; ++kx) {
          int ix = ox * 4 - 1 + kx;
          if ((unsigned)ix >= 8u) continue;
          acc += wc[ky * 6 + kx] * row[iy * 8 + ix];
        }
      }
      float u = acc + bias;
      hb[tid * 4 + oy * 2 + ox] = 0.5f * u * (1.f + erff(u * 0.70710678f));
    }
  __syncthreads();
  if (tid < 8) {
    int o = tid >> 2, pos = tid & 3;
    float acc = 0.f;
    for (int c = 0; c < 64; ++c) acc += w2[o * 64 + c] * hb[c * 4 + pos];
    float off = tanhf(acc) * 4.0f;
    float base = (o == 0) ? (float)(pos & 1) : (float)(pos >> 1);
    vgrid[(size_t)ng * 8 + pos * 2 + o] = 2.f * (base + off) - 1.f;  // normalized
  }
}

// ---- K4: bilinear grid-sample (zero pad) + k/v projections ----------------
__global__ __launch_bounds__(64) void k_kv(const float* __restrict__ xp,
                                           const float* __restrict__ wk,
                                           const float* __restrict__ wvv,
                                           const float* __restrict__ vgrid,
                                           float* __restrict__ kb,
                                           float* __restrict__ vb) {
  __shared__ float img[32 * 64];
  __shared__ float kv[32 * 4];
  __shared__ float vg[8];
  int ng = blockIdx.x;
  int n = ng >> 3, g = ng & 7;
  int tid = threadIdx.x;
  const float* xg = xp + ((size_t)n * DIM_ + g * DPG_) * 64;
  for (int idx = tid; idx < 2048; idx += 64) img[idx] = xg[idx];
  if (tid < 8) vg[tid] = vgrid[(size_t)ng * 8 + tid];
  __syncthreads();
  if (tid < 32) {
    const float* r = img + tid * 64;
    for (int pt = 0; pt < 4; ++pt) {
      float gx = vg[pt * 2], gy = vg[pt * 2 + 1];
      float fx = ((gx + 1.f) * 8.f - 1.f) * 0.5f;
      float fy = ((gy + 1.f) * 8.f - 1.f) * 0.5f;
      float x0 = floorf(fx), y0 = floorf(fy);
      float acc = 0.f;
      for (int dy = 0; dy < 2; ++dy)
        for (int dx = 0; dx < 2; ++dx) {
          float ix = x0 + dx, iy = y0 + dy;
          float wx = dx ? (fx - x0) : (x0 + 1.f - fx);
          float wy = dy ? (fy - y0) : (y0 + 1.f - fy);
          if (ix >= 0.f && ix < 8.f && iy >= 0.f && iy < 8.f)
            acc += wx * wy * r[(int)iy * 8 + (int)ix];
        }
      kv[tid * 4 + pt] = acc;
    }
  }
  __syncthreads();
  const float* wkg = wk + ((size_t)g * 64 + tid) * 32;
  const float* wvg = wvv + ((size_t)g * 64 + tid) * 32;
  for (int pt = 0; pt < 4; ++pt) {
    float ka = 0.f, va = 0.f;
    for (int c = 0; c < 32; ++c) {
      float s = kv[c * 4 + pt];
      ka += wkg[c] * s;
      va += wvg[c] * s;
    }
    kb[((size_t)ng * 4 + pt) * 64 + tid] = ka;   // [ng][j][d]
    vb[((size_t)ng * 4 + pt) * 64 + tid] = va;
  }
}

// ---- K5: CPB MLP (layer2 via WMMA) + softmax + attn@v ---------------------
__global__ __launch_bounds__(128) void k_attn(const float* __restrict__ q,
                                              const float* __restrict__ kbuf,
                                              const float* __restrict__ vbuf,
                                              const float* __restrict__ vgrid,
                                              const float* __restrict__ w1,
                                              const float* __restrict__ b1,
                                              const float* __restrict__ w2,
                                              const float* __restrict__ b2,
                                              const float* __restrict__ w3,
                                              const float* __restrict__ b3,
                                              __bf16* __restrict__ att) {
  __shared__ __align__(16) unsigned char smem[51520];
  __bf16* h1  = (__bf16*)smem;                   // 256 x LD64 bf16 (later: q f32)
  __bf16* w2t = (__bf16*)(smem + 36864);         // 64 x LD64 bf16  (transposed)
  float* kL = (float*)(smem + 46080);            // 4x64
  float* vL = (float*)(smem + 47104);            // 4x64
  float* bz = (float*)(smem + 48128);            // 256 row biases
  float* aL = (float*)(smem + 49152);            // 64x4 attn probs
  float* cw = (float*)(smem + 50176);            // consts (329 f)
  float* cw1 = cw; float* cb1 = cw + 128; float* cb2 = cw + 192;
  float* cw3 = cw + 256; float* cb3 = cw + 320; float* vg = cw + 321;
  int ng = blockIdx.x;
  int n = ng >> 3, g = ng & 7;
  int tid = threadIdx.x;
  cw1[tid] = w1[tid];
  if (tid < 64) { cb1[tid] = b1[tid]; cw3[tid] = w3[tid]; }
  if (tid >= 64 && tid < 128) cb2[tid - 64] = b2[tid - 64];
  if (tid == 0) cb3[0] = b3[0];
  if (tid < 8) vg[tid] = vgrid[(size_t)ng * 8 + tid];
  for (int i = tid; i < 256; i += 128) kL[i] = kbuf[(size_t)ng * 256 + i];
  for (int i = tid; i < 256; i += 128) vL[i] = vbuf[(size_t)ng * 256 + i];
  // cpb_w2 is [k_in][u_out]; store transposed [u][k]
  for (int i4 = tid * 4; i4 < 4096; i4 += 512) {
    float4 v = *(const float4*)(w2 + i4);
    int k = i4 >> 6, u = i4 & 63;
    w2t[(u + 0) * LD64 + k] = (__bf16)v.x;
    w2t[(u + 1) * LD64 + k] = (__bf16)v.y;
    w2t[(u + 2) * LD64 + k] = (__bf16)v.z;
    w2t[(u + 3) * LD64 + k] = (__bf16)v.w;
  }
  __syncthreads();
  // layer 1: pos (2) -> 64, relu, bf16 rows (vectorized b64 stores)
  const float inv7 = 2.f / 7.f;
  for (int rr = tid; rr < 256; rr += 128) {
    int s = rr >> 2, j = rr & 3;
    float qx = (float)(s & 7) * inv7 - 1.f - vg[j * 2];
    float qy = (float)(s >> 3) * inv7 - 1.f - vg[j * 2 + 1];
    float f0 = copysignf(log1pf(fabsf(qx)), qx);
    float f1 = copysignf(log1pf(fabsf(qy)), qy);
    __bf16* hr = h1 + rr * LD64;
    for (int u = 0; u < 64; u += 4) {
      v4bf o;
#pragma unroll
      for (int e = 0; e < 4; ++e) {
        float h = f0 * cw1[u + e] + f1 * cw1[64 + u + e] + cb1[u + e];
        o[e] = (__bf16)fmaxf(h, 0.f);
      }
      *(v4bf*)(hr + u) = o;
    }
  }
  __syncthreads();
  // layer 2 (WMMA, 256x64x64) fused with relu+bias and layer 3 dot(w3)
  int wv_ = tid >> 5, lane = tid & 31;
  for (int tm = 0; tm < 4; ++tm) {
    int mt = wv_ + tm * 4;
    v16bf a0 = ldsA(h1, mt * 16, 0, LD64);
    v16bf a1 = ldsA(h1, mt * 16, 32, LD64);
    float rp[8] = {0, 0, 0, 0, 0, 0, 0, 0};
#pragma unroll
    for (int nt = 0; nt < 4; ++nt) {
      v16bf b0f = ldsBt(w2t, 0, nt * 16, LD64);
      v16bf b1f = ldsBt(w2t, 32, nt * 16, LD64);
      v8f acc; for (int e = 0; e < 8; ++e) acc[e] = 0.f;
      acc = wmma_bf16(a0, b0f, acc);
      acc = wmma_bf16(a1, b1f, acc);
      int u = nt * 16 + (lane & 15);
      float bb = cb2[u], w3v = cw3[u];
#pragma unroll
      for (int r = 0; r < 8; ++r) rp[r] += fmaxf(acc[r] + bb, 0.f) * w3v;
    }
#pragma unroll
    for (int r = 0; r < 8; ++r) {
      float red = rp[r];
      red += __shfl_xor(red, 1, 32);
      red += __shfl_xor(red, 2, 32);
      red += __shfl_xor(red, 4, 32);
      red += __shfl_xor(red, 8, 32);
      if ((lane & 15) == 0) {
        int row = mt * 16 + r + ((lane >> 4) << 3);
        bz[row] = red + cb3[0];
      }
    }
  }
  __syncthreads();
  // reuse h1 region for q tile (f32 16KB)
  float* qs = (float*)smem;
  for (int i = tid; i < 4096; i += 128) qs[i] = q[(size_t)ng * 4096 + i];
  __syncthreads();
  if (tid < 64) {
    int s = tid;
    float lg[4];
    float mx = -1e30f;
    for (int j = 0; j < 4; ++j) {
      float acc = 0.f;
      for (int d = 0; d < 64; ++d) acc += qs[d * 64 + s] * kL[j * 64 + d];
      lg[j] = acc * 0.125f + bz[s * 4 + j];   // SCALE = 1/sqrt(64)
      mx = fmaxf(mx, lg[j]);
    }
    float sum = 0.f;
    for (int j = 0; j < 4; ++j) { lg[j] = __expf(lg[j] - mx); sum += lg[j]; }
    float inv = 1.f / sum;
    for (int j = 0; j < 4; ++j) aL[s * 4 + j] = lg[j] * inv;
  }
  __syncthreads();
  __bf16* ao = att + ((size_t)n * 512 + g * 64) * 64;
  for (int idx = tid; idx < 4096; idx += 128) {
    int d = idx >> 6, s = idx & 63;
    float o = 0.f;
#pragma unroll
    for (int j = 0; j < 4; ++j) o += aL[s * 4 + j] * vL[j * 64 + d];
    ao[d * 64 + s] = (__bf16)o;
  }
}

// ---- K6: output projection 256x512x64 via WMMA + un-patchify --------------
__global__ __launch_bounds__(256) void k_proj(const __bf16* __restrict__ att,
                                              const __bf16* __restrict__ wob,
                                              const float* __restrict__ bo,
                                              float* __restrict__ y) {
  __shared__ __bf16 Ab[256 * LD64];   // wo chunk [M=256][K=64]
  __shared__ __bf16 Bt[64 * LD64];    // att chunk^T [N=s][K=c]
  int n = blockIdx.x;
  int tid = threadIdx.x;
  int wv_ = tid >> 5, lane = tid & 31;
  v8f acc[8];
  for (int t = 0; t < 8; ++t)
    for (int e = 0; e < 8; ++e) acc[t][e] = 0.f;
  const __bf16* an = att + (size_t)n * 512 * 64;
  for (int kc = 0; kc < 8; ++kc) {
    __syncthreads();
    // stage wo chunk: 16B vector copies (already bf16 in global)
    for (int c8 = tid; c8 < 2048; c8 += 256) {
      int m = c8 >> 3, o8 = (c8 & 7) * 8;
      *(v8bf*)(Ab + m * LD64 + o8) =
          *(const v8bf*)(wob + m * 512 + kc * 64 + o8);
    }
    // stage att chunk transposed: [kk][s] -> [s][kk]
    for (int c8 = tid; c8 < 512; c8 += 256) {
      int kk = c8 >> 3, s0 = (c8 & 7) * 8;
      v8bf v = *(const v8bf*)(an + kc * 4096 + kk * 64 + s0);
#pragma unroll
      for (int e = 0; e < 8; ++e) Bt[(s0 + e) * LD64 + kk] = v[e];
    }
    if (kc < 7)  // prefetch next activation chunk into cache hierarchy
      __builtin_prefetch(an + (kc + 1) * 4096 + tid * 16, 0, 1);
    __syncthreads();
    for (int tm = 0; tm < 2; ++tm) {
      int mt = wv_ * 2 + tm;
      v16bf a0 = ldsA(Ab, mt * 16, 0, LD64);
      v16bf a1 = ldsA(Ab, mt * 16, 32, LD64);
#pragma unroll
      for (int nt = 0; nt < 4; ++nt) {
        v16bf b0 = ldsBt(Bt, 0, nt * 16, LD64);
        v16bf b1 = ldsBt(Bt, 32, nt * 16, LD64);
        acc[tm * 4 + nt] = wmma_bf16(a0, b0, acc[tm * 4 + nt]);
        acc[tm * 4 + nt] = wmma_bf16(a1, b1, acc[tm * 4 + nt]);
      }
    }
  }
  int b = n & 15, j = (n >> 4) & 7, i = n >> 7;
  for (int tm = 0; tm < 2; ++tm)
    for (int nt = 0; nt < 4; ++nt) {
      int s = nt * 16 + (lane & 15);
      int p = s >> 3, qq = s & 7;
#pragma unroll
      for (int r = 0; r < 8; ++r) {
        int o = (wv_ * 2 + tm) * 16 + r + ((lane >> 4) << 3);
        float val = acc[tm * 4 + nt][r] + bo[o];
        y[(((size_t)b * 256 + o) * 64 + i * 8 + p) * 64 + j * 8 + qq] = val;
      }
    }
}

// ---- launch ---------------------------------------------------------------
extern "C" void kernel_launch(void* const* d_in, const int* in_sizes, int n_in,
                              void* d_out, int out_size, void* d_ws,
                              size_t ws_size, hipStream_t stream) {
  (void)in_sizes; (void)n_in; (void)out_size; (void)ws_size;
  const float* x   = (const float*)d_in[0];
  const float* lng = (const float*)d_in[1];
  const float* lnb = (const float*)d_in[2];
  const float* wq  = (const float*)d_in[3];
  const float* wk  = (const float*)d_in[4];
  const float* wv  = (const float*)d_in[5];
  const float* ow1 = (const float*)d_in[6];
  const float* ob1 = (const float*)d_in[7];
  const float* ow2 = (const float*)d_in[8];
  const float* cw1 = (const float*)d_in[9];
  const float* cb1 = (const float*)d_in[10];
  const float* cw2 = (const float*)d_in[11];
  const float* cb2 = (const float*)d_in[12];
  const float* cw3 = (const float*)d_in[13];
  const float* cb3 = (const float*)d_in[14];
  const float* wo  = (const float*)d_in[15];
  const float* bo  = (const float*)d_in[16];
  float* y = (float*)d_out;
  char* ws = (char*)d_ws;
  float*  xp    = (float*)(ws);                    //  64 MB
  float*  q     = (float*)(ws + 67108864);         // 128 MB
  float*  vgrid = (float*)(ws + 201326592);        // 256 KB
  float*  kb    = (float*)(ws + 201588736);        //   8 MB
  float*  vb    = (float*)(ws + 209977344);        //   8 MB
  __bf16* att   = (__bf16*)(ws + 218365952);       //  64 MB
  __bf16* wob   = (__bf16*)(ws + 285474816);       // 256 KB

  k_ln  <<<256, 256, 0, stream>>>(x, lng, lnb, xp);
  k_cvt <<<512, 256, 0, stream>>>(wo, wob, 131072);
  k_toq <<<8192, 128, 0, stream>>>(xp, wq, q);
  k_off <<<8192, 64, 0, stream>>>(q, ow1, ob1, ow2, vgrid);
  k_kv  <<<8192, 64, 0, stream>>>(xp, wk, wv, vgrid, kb, vb);
  k_attn<<<8192, 128, 0, stream>>>(q, kb, vb, vgrid, cw1, cb1, cw2, cb2, cw3,
                                   cb3, att);
  k_proj<<<1024, 256, 0, stream>>>(att, wob, bo, y);
}
